// ContextualLoss_forward_39960375722308
// MI455X (gfx1250) — matrix operand order repaired and checked by reading, not compiled, not run on hardware
//
#include <hip/hip_runtime.h>
#include <hip/hip_bf16.h>
#include <math.h>

typedef __attribute__((ext_vector_type(16))) _Float16 v16h;
typedef __attribute__((ext_vector_type(8)))  _Float16 v8h;
typedef __attribute__((ext_vector_type(8)))  float    v8f;

#define BATCH 4
#define CCH   256          // channels (K of the GEMM)
#define MPIX  4096         // 64*64 pixels (M and N of the GEMM)
#define EPSF  2.220446049250313e-16f
#define INV_H 10.0f        // 1 / bandwidth h (h = 0.1)

// -------------------------------------------------------------------------
// zero the per-batch CX accumulators (must happen every launch: harness does
// not re-poison workspace between graph replays)
// -------------------------------------------------------------------------
__global__ void cx_zero_kernel(float* __restrict__ accum) {
  if (threadIdx.x < BATCH) accum[threadIdx.x] = 0.0f;
}

// -------------------------------------------------------------------------
// prep: PONO centering (mean over C of Y), per-pixel channel L2 normalize,
// convert to f16 stored [b][m][c] (c contiguous) so WMMA fragment loads are
// contiguous 16B / 32B global loads.
// One thread per (b, m); channel loads are coalesced across adjacent m.
// -------------------------------------------------------------------------
__global__ void cx_prep_kernel(const float* __restrict__ X,
                               const float* __restrict__ Y,
                               _Float16* __restrict__ Xn,
                               _Float16* __restrict__ Yn) {
  int idx = blockIdx.x * blockDim.x + threadIdx.x;   // idx = b*MPIX + m
  int b = idx / MPIX;
  int m = idx - b * MPIX;
  const float* xb = X + (size_t)b * CCH * MPIX + m;
  const float* yb = Y + (size_t)b * CCH * MPIX + m;

  float mean = 0.0f;
  for (int c = 0; c < CCH; ++c) mean += yb[(size_t)c * MPIX];
  mean *= (1.0f / CCH);

  float sx = 0.0f, sy = 0.0f;
  for (int c = 0; c < CCH; ++c) {
    float xv = xb[(size_t)c * MPIX] - mean;
    float yv = yb[(size_t)c * MPIX] - mean;
    sx += xv * xv;
    sy += yv * yv;
  }
  float inx = 1.0f / (sqrtf(sx) + EPSF);
  float iny = 1.0f / (sqrtf(sy) + EPSF);

  _Float16* xo = Xn + (size_t)idx * CCH;
  _Float16* yo = Yn + (size_t)idx * CCH;
  for (int c = 0; c < CCH; ++c) {
    xo[c] = (_Float16)((xb[(size_t)c * MPIX] - mean) * inx);
    yo[c] = (_Float16)((yb[(size_t)c * MPIX] - mean) * iny);
  }
}

// -------------------------------------------------------------------------
// main pass: per wave, a 16-row strip of S = Xn^T * Yn (K = 256 via 8
// v_wmma_f32_16x16x32_f16 per 16x16 tile). Two passes over the 256 j-tiles:
//   pass 1: running row-min of d = 1 - S
//   pass 2: row-sum of w = exp((1 - d/(min+1e-3)) / h)
// max_j A == w_max / w_sum, with w_max known analytically from the row min.
// 4 waves / block, 64 rows / block, grid = B * (MPIX/64).
// -------------------------------------------------------------------------
__global__ void __launch_bounds__(128)
cx_pass_kernel(const _Float16* __restrict__ Xn,
               const _Float16* __restrict__ Yn,
               float* __restrict__ accum) {
  const int stripsPerBatch = MPIX / 64;
  const int b     = blockIdx.x / stripsPerBatch;
  const int strip = blockIdx.x % stripsPerBatch;
  const int wave  = threadIdx.x >> 5;
  const int lane  = threadIdx.x & 31;
  const int r     = lane & 15;      // row (A) / column (B,C) within tile
  const int kh    = lane >> 4;      // K-half selector
  const int i0    = strip * 64 + wave * 16;

  const _Float16* Xb = Xn + (size_t)b * MPIX * CCH;
  const _Float16* Yb = Yn + (size_t)b * MPIX * CCH;

  // ---- preload the A strip: rows i0..i0+15, all 8 K-tiles (16x32 each) ----
  // A layout: lane L -> row L&15; 16 contiguous K at offset kh*16
  // (matches B).
  v16h afrag[8];
  {
    const _Float16* arow = Xb + (size_t)(i0 + r) * CCH;
#pragma unroll
    for (int kt = 0; kt < 8; ++kt) {
      v8h lo = *(const v8h*)(arow + kt * 32 + kh * 16);
      v8h hi = *(const v8h*)(arow + kt * 32 + kh * 16 + 8);
#pragma unroll
      for (int t = 0; t < 8; ++t) { afrag[kt][t] = lo[t]; afrag[kt][t + 8] = hi[t]; }
    }
  }

  // ---------------- pass 1: row minima of d = 1 - S ----------------
  float rowmin[8];
#pragma unroll
  for (int v = 0; v < 8; ++v) rowmin[v] = 3.402823466e38f;

  for (int jt = 0; jt < MPIX / 16; ++jt) {
    // B layout: lane n -> column n&15, 16 contiguous K at offset kh*16
    const _Float16* brow = Yb + (size_t)(jt * 16 + r) * CCH + kh * 16;
    v8f c = {};
#pragma unroll
    for (int kt = 0; kt < 8; ++kt) {
      v16h bf = *(const v16h*)(brow + kt * 32);
      c = __builtin_amdgcn_wmma_f32_16x16x32_f16(
            /*neg_a=*/false, afrag[kt], /*neg_b=*/false, bf,
            /*c_mod=*/(short)0, c, /*reuse_a=*/false, /*reuse_b=*/false);
    }
#pragma unroll
    for (int v = 0; v < 8; ++v) {
      float d = 1.0f - c[v];
      rowmin[v] = fminf(rowmin[v], d);
    }
  }
  // reduce min across the 16 lanes sharing each row set (bit4 untouched ->
  // lanes 0-15 (rows 0-7) and 16-31 (rows 8-15) stay separate)
#pragma unroll
  for (int v = 0; v < 8; ++v) {
#pragma unroll
    for (int off = 1; off < 16; off <<= 1)
      rowmin[v] = fminf(rowmin[v], __shfl_xor(rowmin[v], off, 32));
  }

  float invden[8], wmax[8];
#pragma unroll
  for (int v = 0; v < 8; ++v) {
    float den = rowmin[v] + 0.001f;
    invden[v] = 1.0f / den;
    wmax[v]   = __expf((1.0f - rowmin[v] * invden[v]) * INV_H);
  }

  // ---------------- pass 2: row sums of w ----------------
  float rowsum[8];
#pragma unroll
  for (int v = 0; v < 8; ++v) rowsum[v] = 0.0f;

  for (int jt = 0; jt < MPIX / 16; ++jt) {
    const _Float16* brow = Yb + (size_t)(jt * 16 + r) * CCH + kh * 16;
    v8f c = {};
#pragma unroll
    for (int kt = 0; kt < 8; ++kt) {
      v16h bf = *(const v16h*)(brow + kt * 32);
      c = __builtin_amdgcn_wmma_f32_16x16x32_f16(
            false, afrag[kt], false, bf, (short)0, c, false, false);
    }
#pragma unroll
    for (int v = 0; v < 8; ++v) {
      float d = 1.0f - c[v];
      rowsum[v] += __expf((1.0f - d * invden[v]) * INV_H);
    }
  }
#pragma unroll
  for (int v = 0; v < 8; ++v) {
#pragma unroll
    for (int off = 1; off < 16; off <<= 1)
      rowsum[v] += __shfl_xor(rowsum[v], off, 32);
  }

  // ---- CX_i = w_max / w_sum; accumulate batch sum (rows are duplicated
  // across 16 lanes -> only the r==0 lane of each half contributes) ----
  if (r == 0) {
    float part = 0.0f;
#pragma unroll
    for (int v = 0; v < 8; ++v) part += wmax[v] / rowsum[v];
    atomicAdd(&accum[b], part);
  }
}

// -------------------------------------------------------------------------
// finalize: loss = mean_b( -log( sum_i CX_i / MPIX ) )
// -------------------------------------------------------------------------
__global__ void cx_final_kernel(const float* __restrict__ accum,
                                float* __restrict__ out) {
  if (threadIdx.x == 0) {
    float s = 0.0f;
    for (int b = 0; b < BATCH; ++b)
      s += -__logf(accum[b] * (1.0f / MPIX));
    out[0] = s * (1.0f / BATCH);
  }
}

extern "C" void kernel_launch(void* const* d_in, const int* in_sizes, int n_in,
                              void* d_out, int out_size, void* d_ws, size_t ws_size,
                              hipStream_t stream) {
  const float* X = (const float*)d_in[0];
  const float* Y = (const float*)d_in[1];
  float* out = (float*)d_out;

  // workspace layout: Xn f16 (8 MB) | Yn f16 (8 MB) | accum (4 floats)
  _Float16* Xn = (_Float16*)d_ws;
  _Float16* Yn = Xn + (size_t)BATCH * MPIX * CCH;
  float* accum = (float*)(Yn + (size_t)BATCH * MPIX * CCH);

  cx_zero_kernel<<<1, 32, 0, stream>>>(accum);
  cx_prep_kernel<<<(BATCH * MPIX) / 256, 256, 0, stream>>>(X, Y, Xn, Yn);
  cx_pass_kernel<<<BATCH * (MPIX / 64), 128, 0, stream>>>(Xn, Yn, accum);
  cx_final_kernel<<<1, 32, 0, stream>>>(accum, out);
}
